// HyperGraphConv_58188216926981
// MI455X (gfx1250) — matrix operand order repaired and checked by reading, not compiled, not run on hardware
//
#include <hip/hip_runtime.h>
#include <math.h>

typedef __attribute__((ext_vector_type(16))) _Float16 v16h;
typedef __attribute__((ext_vector_type(8)))  float    v8f;

#define N_ROWS   20000
#define K_IN     512
#define N_OUT    256
#define DEG      16
#define EPSF     1e-15f
#define MAX_T    (1.0f - 1e-5f)

// ---------- helpers ----------
__device__ __forceinline__ float wsum32(float v) {
#pragma unroll
    for (int off = 16; off > 0; off >>= 1) v += __shfl_xor(v, off, 32);
    return v;
}

__device__ __forceinline__ float artanh_clip(float x) {
    x = fminf(fmaxf(x, -MAX_T), MAX_T);
    return 0.5f * logf((1.0f + x) / (1.0f - x));
}

// ---------- kernel 1: per-row norm of h = 0.25*feat (wave per row) ----------
__global__ void k_xnorm(const float* __restrict__ feat, float* __restrict__ xnorm) {
    int lane = threadIdx.x & 31;
    int row  = blockIdx.x * 8 + (threadIdx.x >> 5);
    float p = 0.f;
#pragma unroll
    for (int j = 0; j < 16; ++j) {
        float v = 0.25f * feat[(size_t)row * K_IN + lane + 32 * j];
        p += v * v;
    }
    float n = sqrtf(wsum32(p));
    if (lane == 0) xnorm[row] = fmaxf(n, EPSF);
}

// ---------- kernel 2: pack A (scaled f16) into WMMA 16x32 fragment layout ----------
// aP[((rowTile*16 + kt)*32 + lane)*16 + i], lane row M = lane&15,
// K = kt*32 + (i/8)*16 + (lane/16)*8 + (i%8)   (ISA 7.12.2, 16-bit A 16x32)
__global__ void k_packA(const float* __restrict__ feat, _Float16* __restrict__ aP) {
    int t    = blockIdx.x * 256 + threadIdx.x;   // t = (rowTile*16 + kt)*32 + lane
    int lane = t & 31;
    int kt   = (t >> 5) & 15;
    int rt   = t >> 9;
    int row  = rt * 16 + (lane & 15);
    int kb   = kt * 32 + ((lane >> 4) << 3);
    _Float16* dst = aP + (size_t)t * 16;
#pragma unroll
    for (int i = 0; i < 16; ++i) {
        int K = kb + ((i >> 3) << 4) + (i & 7);
        dst[i] = (_Float16)(0.25f * feat[(size_t)row * K_IN + K]);
    }
}

// ---------- kernel 3: pack W (f16) into WMMA 32x16 B fragment layout ----------
// wP[((nt*16 + kt)*32 + lane)*16 + i], lane col N = nt*16 + (lane&15),
// K = kt*32 + (i/8)*16 + (lane/16)*8 + (i%8)
__global__ void k_packW(const float* __restrict__ W, _Float16* __restrict__ wP) {
    int t    = blockIdx.x * 256 + threadIdx.x;   // 8192 total
    int lane = t & 31;
    int kt   = (t >> 5) & 15;
    int nt   = t >> 9;
    int n    = nt * 16 + (lane & 15);
    int kb   = kt * 32 + ((lane >> 4) << 3);
    _Float16* dst = wP + (size_t)t * 16;
#pragma unroll
    for (int i = 0; i < 16; ++i)
        dst[i] = (_Float16)W[(size_t)(kb + ((i >> 3) << 4) + (i & 7)) * N_OUT + n];
}

// ---------- kernel 4: WMMA GEMM + fused mobius_matvec epilogue ----------
// block = 256 threads (8 waves), one 16-row M-tile per block, full 256 cols.
// wave w computes N-tiles 2w and 2w+1; 16 K-steps of v_wmma_f32_16x16x32_f16.
__global__ void k_gemm(const _Float16* __restrict__ aP,
                       const _Float16* __restrict__ wP,
                       const float* __restrict__ xnorm,
                       float* __restrict__ h2) {
    __shared__ float s_sum[16];
    __shared__ float s_scale[16];
    const int lane = threadIdx.x & 31;
    const int wave = threadIdx.x >> 5;
    const int mt   = blockIdx.x;

    if (threadIdx.x < 16) s_sum[threadIdx.x] = 0.f;
    __syncthreads();

    v8f c0 = {}, c1 = {};
    const int nt0 = wave * 2, nt1 = wave * 2 + 1;
#pragma unroll
    for (int kt = 0; kt < 16; ++kt) {
        v16h a  = *(const v16h*)(aP + ((size_t)(mt * 16 + kt) * 32 + lane) * 16);
        v16h b0 = *(const v16h*)(wP + ((size_t)(nt0 * 16 + kt) * 32 + lane) * 16);
        v16h b1 = *(const v16h*)(wP + ((size_t)(nt1 * 16 + kt) * 32 + lane) * 16);
        c0 = __builtin_amdgcn_wmma_f32_16x16x32_f16(false, a, false, b0,
                                                    (short)0, c0, false, false);
        c1 = __builtin_amdgcn_wmma_f32_16x16x32_f16(false, a, false, b1,
                                                    (short)0, c1, false, false);
    }

    // C/D layout: VGPR r, lane L -> row m = r + 8*(L/16), col n = L%16
    const int mbase = (lane >> 4) * 8;
    const int ncol  = lane & 15;
#pragma unroll
    for (int r = 0; r < 8; ++r)
        atomicAdd(&s_sum[mbase + r], c0[r] * c0[r] + c1[r] * c1[r]);
    __syncthreads();

    if (threadIdx.x < 16) {
        int   row = mt * 16 + threadIdx.x;
        float mxn = fmaxf(sqrtf(s_sum[threadIdx.x]), EPSF);
        float xn  = xnorm[row];                       // >= EPS already
        float t   = tanhf((mxn / xn) * artanh_clip(xn));
        s_scale[threadIdx.x] = (mxn <= 1e-10f) ? 0.f : (t / mxn);
    }
    __syncthreads();

#pragma unroll
    for (int r = 0; r < 8; ++r) {
        int   m   = mbase + r;
        float sc  = s_scale[m];
        size_t row = (size_t)(mt * 16 + m);
        h2[row * N_OUT + nt0 * 16 + ncol] = c0[r] * sc;
        h2[row * N_OUT + nt1 * 16 + ncol] = c1[r] * sc;
    }
}

// ---------- kernel 5: gather + sequential mobius_add scan + epilogue ----------
// wave per node; lane owns dims {lane + 32*j}, j=0..7 (coalesced 256-wide rows)
__device__ __forceinline__ void mobius_add8(float* __restrict__ x,
                                            const float* __restrict__ y) {
    float px = 0.f, py = 0.f, pxy = 0.f;
#pragma unroll
    for (int j = 0; j < 8; ++j) {
        px  += x[j] * x[j];
        py  += y[j] * y[j];
        pxy += x[j] * y[j];
    }
    float x2 = wsum32(px), y2 = wsum32(py), xy = wsum32(pxy);
    float a   = 1.f + 2.f * xy + y2;
    float b   = 1.f - x2;
    float den = fmaxf(1.f + 2.f * xy + x2 * y2, EPSF);
    float inv = 1.f / den;
#pragma unroll
    for (int j = 0; j < 8; ++j) x[j] = (a * x[j] + b * y[j]) * inv;
}

__global__ void k_agg(const float* __restrict__ h2,
                      const float* __restrict__ bias,
                      const int* __restrict__ src,
                      float* __restrict__ out) {
    const int lane = threadIdx.x & 31;
    const int node = blockIdx.x * 8 + (threadIdx.x >> 5);
    const int* s = src + (size_t)node * DEG;

    float bv[8], x[8], y[8];
#pragma unroll
    for (int j = 0; j < 8; ++j) bv[j] = bias[lane + 32 * j];

    size_t r0 = (size_t)s[0] * N_OUT;
#pragma unroll
    for (int j = 0; j < 8; ++j) x[j] = h2[r0 + lane + 32 * j];

    for (int e = 1; e < DEG; ++e) {
        size_t ri = (size_t)s[e] * N_OUT;
        if (e + 1 < DEG)  // prefetch next gathered row (global_prefetch_b8)
            __builtin_prefetch(h2 + (size_t)s[e + 1] * N_OUT + lane, 0, 0);
#pragma unroll
        for (int j = 0; j < 8; ++j) y[j] = h2[ri + lane + 32 * j];
        mobius_add8(x, y);
    }

    // rst * deg^-1/2, then mobius_add with bias
#pragma unroll
    for (int j = 0; j < 8; ++j) x[j] *= 0.25f;
    mobius_add8(x, bv);

    // expmap0(relu(logmap0(rst)))
    float p = 0.f;
#pragma unroll
    for (int j = 0; j < 8; ++j) p += x[j] * x[j];
    float yn = fmaxf(sqrtf(wsum32(p)), EPSF);
    float s1 = artanh_clip(yn) / yn;

    float u[8];
    float q = 0.f;
#pragma unroll
    for (int j = 0; j < 8; ++j) {
        u[j] = fmaxf(s1 * x[j], 0.f);
        q   += u[j] * u[j];
    }
    float un = fmaxf(sqrtf(wsum32(q)), EPSF);
    float s2 = tanhf(un) / un;

    size_t ob = (size_t)node * N_OUT;
#pragma unroll
    for (int j = 0; j < 8; ++j) out[ob + lane + 32 * j] = s2 * u[j];
}

// ---------- launch ----------
extern "C" void kernel_launch(void* const* d_in, const int* in_sizes, int n_in,
                              void* d_out, int out_size, void* d_ws, size_t ws_size,
                              hipStream_t stream) {
    const float* feat   = (const float*)d_in[0];   // 20000*512
    const float* weight = (const float*)d_in[1];   // 512*256
    const float* bias   = (const float*)d_in[2];   // 256
    const int*   src    = (const int*)d_in[3];     // 20000*16
    float*       out    = (float*)d_out;           // 20000*256

    char* ws = (char*)d_ws;
    // layout (bytes, 256-aligned):
    //   aP    : 20000*512*2 = 20,480,000
    //   wP    :   512*256*2 =    262,144
    //   xnorm :  20000*4    =     80,000 (padded to 80,128)
    //   h2    : 20000*256*4 = 20,480,000
    _Float16* aP    = (_Float16*)(ws);
    _Float16* wP    = (_Float16*)(ws + 20480000);
    float*    xnorm = (float*)   (ws + 20742144);
    float*    h2    = (float*)   (ws + 20822272);

    k_xnorm<<<2500, 256, 0, stream>>>(feat, xnorm);
    k_packA<<<2500, 256, 0, stream>>>(feat, aP);    // 1250*16*32 lane-entries
    k_packW<<<32,   256, 0, stream>>>(weight, wP);  // 16*16*32 lane-entries
    k_gemm <<<1250, 256, 0, stream>>>(aP, wP, xnorm, h2);
    k_agg  <<<2500, 256, 0, stream>>>(h2, bias, src, out);
}
